// TensorReductionLayer_15161234555080
// MI455X (gfx1250) — matrix operand order repaired and checked by reading, not compiled, not run on hardware
//
#include <hip/hip_runtime.h>

// ---------------------------------------------------------------------------
// out[b,k] = sum_{i,j} x[b,i] x[b,j] W[i,j,k]  as a single GEMM:
//   C[2048,512] = A[2048,65536] * W[65536,512],  A[b, i*256+j] = x[b,i]*x[b,j]
// A is built in registers from an LDS-resident bf16 x-tile (TDM-staged).
// W is pre-converted to bf16 in a WMMA-fragment-ready layout in d_ws.
// Reduction split 4-way over i (grid.z) into f32 partials + reduce kernel.
// Matrix path: v_wmma_f32_16x16x32_bf16 (f32 accumulate).
// Workspace: 64 MB Wb + 1 MB Xb + 16 MB partials = 81 MB.
// ---------------------------------------------------------------------------

typedef __bf16 bf16;
typedef __attribute__((ext_vector_type(16))) __bf16 v16bf;
typedef __attribute__((ext_vector_type(8)))  __bf16 v8bf;
typedef __attribute__((ext_vector_type(8)))  float  v8f;
typedef __attribute__((ext_vector_type(4)))  unsigned int v4u;
typedef __attribute__((ext_vector_type(8)))  int    v8i;
typedef __attribute__((ext_vector_type(4)))  int    v4i;

#define BATCH   2048
#define IN_F    256
#define OUT_F   512
#define KTOT    (IN_F * IN_F)      // 65536
#define KTILES  (KTOT / 32)        // 2048 K-slabs
#define BM      128
#define BN      64
#define NSPLIT  4                  // i-dimension split (grid.z)
#define ISPLIT  (IN_F / NSPLIT)    // 64 i-values per split
#define XPITCH  264                // 256 + 8 bf16 (16B) row pad in LDS
#define WB_ELEMS (KTOT * OUT_F)    // 33,554,432 bf16
#define XB_ELEMS (BATCH * IN_F)    // 524,288 bf16
#define OUT_ELEMS (BATCH * OUT_F)  // 1,048,576 f32

// ---------------------------------------------------------------------------
// Kernel 1: convert W f32 -> bf16 fragment-ready tiles (ISA 16-bit B layout):
//   tile (kt,nt): K=kt*32..+31, N=nt*16..+15; elem = (nt*KTILES+kt)*512
//                 + g*256 + lane*8 + hh ; lane=((K&8)?16:0)|n_lo ; K=(lane&16)/2+g*16+hh
// Also converts x f32 -> bf16 row-major (blocks >= 16384).
// ---------------------------------------------------------------------------
__global__ __launch_bounds__(256) void convert_kernel(
        const float* __restrict__ W, const float* __restrict__ X,
        bf16* __restrict__ Wb, bf16* __restrict__ Xb) {
    if (blockIdx.x < 16384) {
        unsigned t        = blockIdx.x * 256u + threadIdx.x;
        unsigned elemBase = t * 8u;
        unsigned tileIdx  = elemBase >> 9;
        unsigned rem      = elemBase & 511u;
        unsigned g        = rem >> 8;
        unsigned lane     = (rem >> 3) & 31u;
        unsigned nt       = tileIdx >> 11;
        unsigned kt       = tileIdx & 2047u;
        unsigned n        = nt * 16u + (lane & 15u);
        unsigned kbase    = kt * 32u + ((lane & 16u) >> 1) + g * 16u;
        v8bf o;
#pragma unroll
        for (int hh = 0; hh < 8; ++hh)
            o[hh] = (bf16)W[(kbase + hh) * OUT_F + n];
        *(v8bf*)(Wb + elemBase) = o;
    } else {
        unsigned t = (blockIdx.x - 16384u) * 256u + threadIdx.x;
        unsigned e = t * 8u;                      // < 524288
        v8bf o;
#pragma unroll
        for (int hh = 0; hh < 8; ++hh)
            o[hh] = (bf16)X[e + hh];
        *(v8bf*)(Xb + e) = o;
    }
}

// ---------------------------------------------------------------------------
// Kernel 2: WMMA GEMM. Block = 256 threads = 8 waves (4M x 2N),
// wave tile 32x32 (2x2 accumulators), block tile 128x64, grid.z = i-split.
// x-tile staged into LDS by the Tensor Data Mover (hardware row padding).
// ---------------------------------------------------------------------------
__global__ __launch_bounds__(256) void bilinear_wmma_kernel(
        const bf16* __restrict__ Xb, const bf16* __restrict__ Wb,
        float* __restrict__ Part) {
    __shared__ bf16 xt[BM * XPITCH];

    const int tid   = threadIdx.x;
    const int lane  = tid & 31;
    const int wave  = tid >> 5;
    const int waveM = wave & 3;
    const int waveN = wave >> 2;
    const int mBase = blockIdx.x * BM;
    const int nBase = blockIdx.y * BN;
    const int i0    = blockIdx.z * ISPLIT;

#if __has_builtin(__builtin_amdgcn_tensor_load_to_lds)
    // --- TDM: DMA the 128x256 bf16 x-tile into LDS with 16B row padding ---
    if (wave == 0) {
        unsigned ldsOff = (unsigned)(unsigned long long)(void*)&xt[0];
        unsigned long long ga =
            (unsigned long long)(const void*)(Xb + (unsigned)mBase * IN_F);
        v4u g0;
        g0[0] = 1u;                                   // count=1, user mode
        g0[1] = ldsOff;                               // lds_addr (bytes)
        g0[2] = (unsigned)ga;                         // global_addr[31:0]
        g0[3] = (unsigned)((ga >> 32) & 0x1FFFFFFu)   // global_addr[56:32]
              | (2u << 30);                           // type = 2 (image)
        v8i g1;
        g1[0] = (int)((1u << 16)      // data_size = 2 bytes
                    | (1u << 20)      // pad_enable
                    | (6u << 22)      // pad_interval: 128 DWORDs (one row)
                    | (3u << 25));    // pad_amount: 4 DWORDs (16 B)
        g1[1] = (int)(256u << 16);    // tensor_dim0 = 256 (bits 79:48 lo)
        g1[2] = (int)(2048u << 16);   // tensor_dim1 = 2048 (bits 111:80 lo)
        g1[3] = (int)(256u << 16);    // tile_dim0 = 256 (bits 127:112)
        g1[4] = 128;                  // tile_dim1 = 128 (bits 143:128)
        g1[5] = 256;                  // tensor_dim0_stride = 256 (bits 207:160)
        g1[6] = 0;
        g1[7] = 0;
        v4i gz4; gz4[0] = 0; gz4[1] = 0; gz4[2] = 0; gz4[3] = 0;
        v8i gz8;
#pragma unroll
        for (int q = 0; q < 8; ++q) gz8[q] = 0;
        // 6-arg form (clang-23 / therock headers): groups 2/3 + extra group
        // are unused (zero) for a 2D tensor descriptor.
        __builtin_amdgcn_tensor_load_to_lds(g0, g1, gz4, gz4, gz8, 0);
        __builtin_amdgcn_s_wait_tensorcnt(0);
    }
    __syncthreads();
#else
    for (int idx = tid; idx < BM * IN_F; idx += 256) {
        int r = idx >> 8, c = idx & 255;
        xt[r * XPITCH + c] = Xb[(mBase + r) * IN_F + c];
    }
    __syncthreads();
#endif

    // A-fragment addressing (ISA 16-bit A 16x32 layout)
    const int row0 = waveM * 32 + (lane & 15);
    const int row1 = row0 + 16;
    const int kbA  = (lane >> 4) * 8;
    const int nt0  = (nBase >> 4) + waveN * 2;

    v8f zero;
#pragma unroll
    for (int q = 0; q < 8; ++q) zero[q] = 0.0f;
    v8f acc[2][2];
#pragma unroll
    for (int mi = 0; mi < 2; ++mi)
#pragma unroll
        for (int ni = 0; ni < 2; ++ni) acc[mi][ni] = zero;

    union V16 { v16bf v; v8bf h[2]; };

    for (int jt = 0; jt < 8; ++jt) {
        const int j0 = jt * 32;
        V16 xa[2];                       // i-invariant x j-slices in VGPRs
        xa[0].h[0] = *(const v8bf*)&xt[row0 * XPITCH + j0 + kbA];
        xa[0].h[1] = *(const v8bf*)&xt[row0 * XPITCH + j0 + kbA + 16];
        xa[1].h[0] = *(const v8bf*)&xt[row1 * XPITCH + j0 + kbA];
        xa[1].h[1] = *(const v8bf*)&xt[row1 * XPITCH + j0 + kbA + 16];

#pragma unroll 2
        for (int i = i0; i < i0 + ISPLIT; ++i) {
            const int kt = i * 8 + jt;
            V16 bfrag[2];
#pragma unroll
            for (int ni = 0; ni < 2; ++ni) {
                const bf16* tb = Wb + (((nt0 + ni) * KTILES + kt) << 9);
                bfrag[ni].h[0] = *(const v8bf*)(tb + lane * 8);
                bfrag[ni].h[1] = *(const v8bf*)(tb + 256 + lane * 8);
                __builtin_prefetch((const void*)(tb + (8 << 9) + lane * 8),
                                   0, 1);
            }
            bf16 s0 = xt[row0 * XPITCH + i];
            bf16 s1 = xt[row1 * XPITCH + i];
            v16bf vs0, vs1;
#pragma unroll
            for (int e = 0; e < 16; ++e) { vs0[e] = s0; vs1[e] = s1; }
            v16bf a0 = xa[0].v * vs0;
            v16bf a1 = xa[1].v * vs1;
#pragma unroll
            for (int ni = 0; ni < 2; ++ni) {
                acc[0][ni] = __builtin_amdgcn_wmma_f32_16x16x32_bf16(
                    false, a0, false, bfrag[ni].v, (short)0, acc[0][ni],
                    false, false);
                acc[1][ni] = __builtin_amdgcn_wmma_f32_16x16x32_bf16(
                    false, a1, false, bfrag[ni].v, (short)0, acc[1][ni],
                    false, false);
            }
        }
    }

    // Store partials per ISA 16x16 f32 D layout (lanes 0-15:M=v, 16-31:M=v+8)
    float* P = Part + (unsigned)blockIdx.z * OUT_ELEMS;
    const int rHi  = (lane >> 4) * 8;
    const int nCol = lane & 15;
#pragma unroll
    for (int mi = 0; mi < 2; ++mi)
#pragma unroll
        for (int ni = 0; ni < 2; ++ni) {
            const int col = nBase + waveN * 32 + ni * 16 + nCol;
#pragma unroll
            for (int v = 0; v < 8; ++v) {
                const int row = mBase + waveM * 32 + mi * 16 + rHi + v;
                P[row * OUT_F + col] = acc[mi][ni][v];
            }
        }
}

// ---------------------------------------------------------------------------
// Kernel 3: sum the 4 split partials into the final output (vectorized).
// ---------------------------------------------------------------------------
__global__ __launch_bounds__(256) void reduce_kernel(
        const float* __restrict__ Part, float* __restrict__ Out) {
    unsigned e = (blockIdx.x * 256u + threadIdx.x) * 4u;   // < 1,048,576
    float4 a = *(const float4*)(Part + e);
    float4 b = *(const float4*)(Part + e + OUT_ELEMS);
    float4 c = *(const float4*)(Part + e + 2u * OUT_ELEMS);
    float4 d = *(const float4*)(Part + e + 3u * OUT_ELEMS);
    float4 r;
    r.x = a.x + b.x + c.x + d.x;
    r.y = a.y + b.y + c.y + d.y;
    r.z = a.z + b.z + c.z + d.z;
    r.w = a.w + b.w + c.w + d.w;
    *(float4*)(Out + e) = r;
}

// ---------------------------------------------------------------------------
extern "C" void kernel_launch(void* const* d_in, const int* in_sizes, int n_in,
                              void* d_out, int out_size, void* d_ws,
                              size_t ws_size, hipStream_t stream) {
    const float* x = (const float*)d_in[0];   // [2048, 256]
    const float* W = (const float*)d_in[1];   // [256, 256, 512]
    float* out     = (float*)d_out;           // [2048, 512]

    bf16*  Wb   = (bf16*)d_ws;                         // 64 MB
    bf16*  Xb   = Wb + WB_ELEMS;                       // 1 MB
    float* Part = (float*)(Xb + XB_ELEMS);             // 16 MB (4 partials)

    // W swizzle+convert (16384 blocks) and x convert (256 blocks).
    convert_kernel<<<16640, 256, 0, stream>>>(W, x, Wb, Xb);

    dim3 grid(BATCH / BM, OUT_F / BN, NSPLIT);         // (16, 8, 4)
    bilinear_wmma_kernel<<<grid, 256, 0, stream>>>(Xb, Wb, Part);

    reduce_kernel<<<OUT_ELEMS / 4 / 256, 256, 0, stream>>>(Part, out);
}
